// RelGraphConv_10385230921813
// MI455X (gfx1250) — compile-verified, matrix operand order checked by reading, not run on hardware
//
#include <hip/hip_runtime.h>

// ---------------------------------------------------------------------------
// RelGraphConv for MI455X (gfx1250, wave32, WMMA).
//   m = concat([h+r, h*r, h, r]) @ Wn   (per edge, K=256, N=64)  -> bf16 WMMA
//   s = segment_sum(m, dst); cnt = segment_sum(1, dst)
//   out = s / max(cnt,1) + feat @ loop_weight                     -> bf16 WMMA
// ---------------------------------------------------------------------------

typedef __bf16 bf16_t;
typedef __attribute__((ext_vector_type(4)))  __bf16 v4bf;
typedef __attribute__((ext_vector_type(8)))  __bf16 v8bf;
typedef __attribute__((ext_vector_type(16))) __bf16 v16bf;
typedef __attribute__((ext_vector_type(8)))  float  v8f;

#define IN_FEAT  64
#define OUT_FEAT 64
#define CAT_K    256
#define APAD     264   // 256 + 8 bf16 pad: row = 528B = 33*16B (b128-aligned, bank-skewed)
#define WPAD     264
#define LPAD     72    // 64 + 8 bf16 pad: row = 144B = 9*16B

static __device__ __forceinline__ v16bf cat8(v8bf lo, v8bf hi) {
  return __builtin_shufflevector(lo, hi, 0,1,2,3,4,5,6,7,8,9,10,11,12,13,14,15);
}

// --------------------------------- zero ------------------------------------
__global__ void rgcn_zero_kernel(float* __restrict__ p, long n) {
  long i = (long)blockIdx.x * blockDim.x + threadIdx.x;
  if (i < n) p[i] = 0.0f;
}

// ------------------------------ edge WMMA ----------------------------------
// 128 threads = 4 waves. Block processes 16 edges per iteration; wave w owns
// output columns [16w, 16w+16). B fragments (weight) hoisted to registers.
__global__ __launch_bounds__(128) void rgcn_edge_wmma(
    const float* __restrict__ feat, const float* __restrict__ rel,
    const float* __restrict__ Wn,   const int* __restrict__ src,
    const int* __restrict__ dst,    const int* __restrict__ etype,
    float* __restrict__ ssum, float* __restrict__ cnt, int ntiles)
{
  __shared__ alignas(16) bf16_t Wt[OUT_FEAT][WPAD];   // Wt[n][k] = Wn[k][n]
  __shared__ alignas(16) bf16_t Asta[16][APAD];       // 16 edges x 256 concat feats
  __shared__ int dsh[16];

  const int tid  = threadIdx.x;
  const int lane = tid & 31;
  const int wv   = tid >> 5;

  // Stage weight (transposed) into LDS as bf16 once per block.
  for (int i = tid; i < CAT_K * OUT_FEAT; i += 128) {
    int k = i >> 6, n = i & 63;
    Wt[n][k] = (bf16_t)Wn[i];
  }
  __syncthreads();

  // Hoist this wave's 8 B fragments (one per K-step of 32) into registers.
  // 16-bit B layout: lane&15 -> column n; lanes 16-31 hold K+8 / K+24 halves.
  const int nb = lane & 15;
  const int kh = (lane >> 4) << 3;           // 0 or 8
  const int ncol = wv * 16 + nb;
  v16bf bfrag[8];
#pragma unroll
  for (int kb = 0; kb < 8; ++kb) {
    const int k0 = kb * 32;
    v8bf blo = *(const v8bf*)&Wt[ncol][k0 + kh];
    v8bf bhi = *(const v8bf*)&Wt[ncol][k0 + 16 + kh];
    bfrag[kb] = cat8(blo, bhi);
  }

  for (int g = blockIdx.x; g < ntiles; g += gridDim.x) {
    const int ebase = g * 16;

    { // ---- gather: 8 threads per edge, 8 features each ----
      const int el   = tid >> 3;             // edge slot 0..15
      const int part = tid & 7;
      const int e    = ebase + el;
      const int sv   = src[e];
      const int ev   = etype[e];
      if (part == 0) {
        int dv = dst[e];
        dsh[el] = dv;
        atomicAdd(&cnt[dv], 1.0f);
      }
      bf16_t* Arow = Asta[el];
      const int f0 = part * 8;
      const float4* hp = (const float4*)(feat + (long)sv * IN_FEAT + f0);
      const float4* rp = (const float4*)(rel  + (long)ev * IN_FEAT + f0);
#pragma unroll
      for (int q = 0; q < 2; ++q) {
        float4 h4 = hp[q];
        float4 r4 = rp[q];
        v4bf vs = { (bf16_t)(h4.x + r4.x), (bf16_t)(h4.y + r4.y),
                    (bf16_t)(h4.z + r4.z), (bf16_t)(h4.w + r4.w) };
        v4bf vp = { (bf16_t)(h4.x * r4.x), (bf16_t)(h4.y * r4.y),
                    (bf16_t)(h4.z * r4.z), (bf16_t)(h4.w * r4.w) };
        v4bf vh = { (bf16_t)h4.x, (bf16_t)h4.y, (bf16_t)h4.z, (bf16_t)h4.w };
        v4bf vr = { (bf16_t)r4.x, (bf16_t)r4.y, (bf16_t)r4.z, (bf16_t)r4.w };
        const int f = f0 + q * 4;
        *(v4bf*)&Arow[      f] = vs;
        *(v4bf*)&Arow[ 64 + f] = vp;
        *(v4bf*)&Arow[128 + f] = vh;
        *(v4bf*)&Arow[192 + f] = vr;
      }
    }
    __syncthreads();

    // ---- 16x16x256 slice via 8 chained bf16 WMMAs (wave-uniform, EXEC=all) ----
    v8f acc = {};
    const int mA = lane & 15;
#pragma unroll
    for (int kb = 0; kb < 8; ++kb) {
      const int k0 = kb * 32;
      v8bf alo = *(const v8bf*)&Asta[mA][k0 + kh];
      v8bf ahi = *(const v8bf*)&Asta[mA][k0 + 16 + kh];
      v16bf a = cat8(alo, ahi);
      acc = __builtin_amdgcn_wmma_f32_16x16x32_bf16(
          false, a, false, bfrag[kb], (short)0, acc, false, false);
    }

    // ---- scatter-add to destination rows (C layout: vgpr j -> M=j or j+8) ----
    const int hi8 = (lane >> 4) << 3;
#pragma unroll
    for (int j = 0; j < 8; ++j) {
      const int m   = j + hi8;
      const int row = dsh[m];
      atomicAdd(&ssum[(long)row * OUT_FEAT + ncol], acc[j]);
    }
    __syncthreads();
  }
}

// scalar tail for n_edges % 16 != 0 (unused for 800000 edges)
__global__ void rgcn_edge_tail(
    const float* __restrict__ feat, const float* __restrict__ rel,
    const float* __restrict__ Wn,   const int* __restrict__ src,
    const int* __restrict__ dst,    const int* __restrict__ etype,
    float* __restrict__ ssum, float* __restrict__ cnt, int ebase, int n_edges)
{
  const int e = ebase + blockIdx.x;
  if (e >= n_edges) return;
  __shared__ float catv[CAT_K];
  const int f  = threadIdx.x;            // 64 threads
  const int sv = src[e], ev = etype[e];
  float h = feat[(long)sv * IN_FEAT + f];
  float r = rel[(long)ev * IN_FEAT + f];
  catv[f] = h + r; catv[64 + f] = h * r; catv[128 + f] = h; catv[192 + f] = r;
  __syncthreads();
  float acc = 0.0f;
  for (int k = 0; k < CAT_K; ++k) acc += catv[k] * Wn[k * OUT_FEAT + f];
  const int dv = dst[e];
  atomicAdd(&ssum[(long)dv * OUT_FEAT + f], acc);
  if (f == 0) atomicAdd(&cnt[dv], 1.0f);
}

// ------------------------------ final WMMA ---------------------------------
// out[n,:] = ssum[n,:]/max(cnt,1) + (feat @ loop_weight)[n,:]
__global__ __launch_bounds__(128) void rgcn_final_wmma(
    const float* __restrict__ feat, const float* __restrict__ LW,
    const float* __restrict__ ssum, const float* __restrict__ cnt,
    float* __restrict__ out, int ntiles)
{
  __shared__ alignas(16) bf16_t LWt[OUT_FEAT][LPAD];  // LWt[n][k]
  __shared__ alignas(16) bf16_t Fst[16][LPAD];

  const int tid  = threadIdx.x;
  const int lane = tid & 31;
  const int wv   = tid >> 5;

  for (int i = tid; i < IN_FEAT * OUT_FEAT; i += 128) {
    int k = i >> 6, n = i & 63;
    LWt[n][k] = (bf16_t)LW[i];
  }
  __syncthreads();

  const int nb = lane & 15;
  const int kh = (lane >> 4) << 3;
  const int ncol = wv * 16 + nb;
  v16bf bfrag[2];
#pragma unroll
  for (int kb = 0; kb < 2; ++kb) {
    const int k0 = kb * 32;
    v8bf blo = *(const v8bf*)&LWt[ncol][k0 + kh];
    v8bf bhi = *(const v8bf*)&LWt[ncol][k0 + 16 + kh];
    bfrag[kb] = cat8(blo, bhi);
  }

  for (int g = blockIdx.x; g < ntiles; g += gridDim.x) {
    const int base = g * 16;
    { // gather 16 node rows as bf16
      const int el   = tid >> 3;
      const int part = tid & 7;
      const int node = base + el;
      const int f0   = part * 8;
      const float4* fp = (const float4*)(feat + (long)node * IN_FEAT + f0);
      bf16_t* Frow = Fst[el];
#pragma unroll
      for (int q = 0; q < 2; ++q) {
        float4 v = fp[q];
        v4bf pv = { (bf16_t)v.x, (bf16_t)v.y, (bf16_t)v.z, (bf16_t)v.w };
        *(v4bf*)&Frow[f0 + q * 4] = pv;
      }
    }
    __syncthreads();

    v8f acc = {};
    const int mA = lane & 15;
#pragma unroll
    for (int kb = 0; kb < 2; ++kb) {
      const int k0 = kb * 32;
      v8bf alo = *(const v8bf*)&Fst[mA][k0 + kh];
      v8bf ahi = *(const v8bf*)&Fst[mA][k0 + 16 + kh];
      v16bf a = cat8(alo, ahi);
      acc = __builtin_amdgcn_wmma_f32_16x16x32_bf16(
          false, a, false, bfrag[kb], (short)0, acc, false, false);
    }

    const int hi8 = (lane >> 4) << 3;
#pragma unroll
    for (int j = 0; j < 8; ++j) {
      const int node = base + j + hi8;
      const float sc = 1.0f / fmaxf(cnt[node], 1.0f);
      out[(long)node * OUT_FEAT + ncol] =
          ssum[(long)node * OUT_FEAT + ncol] * sc + acc[j];
    }
    __syncthreads();
  }
}

// scalar tail for n_nodes % 16 != 0 (unused for 50000 nodes)
__global__ void rgcn_final_tail(
    const float* __restrict__ feat, const float* __restrict__ LW,
    const float* __restrict__ ssum, const float* __restrict__ cnt,
    float* __restrict__ out, int nbase, int n_nodes)
{
  const int node = nbase + blockIdx.x;
  if (node >= n_nodes) return;
  const int f = threadIdx.x;             // 64 threads
  float acc = 0.0f;
  for (int k = 0; k < IN_FEAT; ++k)
    acc += feat[(long)node * IN_FEAT + k] * LW[k * OUT_FEAT + f];
  const float sc = 1.0f / fmaxf(cnt[node], 1.0f);
  out[(long)node * OUT_FEAT + f] = ssum[(long)node * OUT_FEAT + f] * sc + acc;
}

// ------------------------------- launcher ----------------------------------
extern "C" void kernel_launch(void* const* d_in, const int* in_sizes, int n_in,
                              void* d_out, int out_size, void* d_ws, size_t ws_size,
                              hipStream_t stream) {
  const float* feat = (const float*)d_in[0];
  const float* rel  = (const float*)d_in[1];
  const float* Wn   = (const float*)d_in[2];
  const float* LW   = (const float*)d_in[3];
  const int*   src  = (const int*)d_in[4];
  const int*   dst  = (const int*)d_in[5];
  const int*   ety  = (const int*)d_in[6];
  float* out = (float*)d_out;

  const int n_nodes = in_sizes[0] / IN_FEAT;
  const int n_edges = in_sizes[4];

  float* ssum = (float*)d_ws;                          // [n_nodes, 64]
  float* cnt  = ssum + (long)n_nodes * OUT_FEAT;       // [n_nodes]

  // 1) zero accumulators (must run every call; ws is poisoned once)
  const long nz = (long)n_nodes * OUT_FEAT + n_nodes;
  rgcn_zero_kernel<<<(int)((nz + 255) / 256), 256, 0, stream>>>(ssum, nz);

  // 2) edge messages + scatter
  const int etiles = n_edges / 16;
  if (etiles > 0) {
    int blocks = etiles < 2048 ? etiles : 2048;
    rgcn_edge_wmma<<<blocks, 128, 0, stream>>>(feat, rel, Wn, src, dst, ety,
                                               ssum, cnt, etiles);
  }
  const int etail = n_edges - etiles * 16;
  if (etail > 0)
    rgcn_edge_tail<<<etail, 64, 0, stream>>>(feat, rel, Wn, src, dst, ety,
                                             ssum, cnt, etiles * 16, n_edges);

  // 3) mean + self-loop matmul, fused
  const int ftiles = n_nodes / 16;
  if (ftiles > 0) {
    int blocks = ftiles < 2048 ? ftiles : 2048;
    rgcn_final_wmma<<<blocks, 128, 0, stream>>>(feat, LW, ssum, cnt, out, ftiles);
  }
  const int ntail = n_nodes - ftiles * 16;
  if (ntail > 0)
    rgcn_final_tail<<<ntail, 64, 0, stream>>>(feat, LW, ssum, cnt, out,
                                              ftiles * 16, n_nodes);
}
